// Decoder_32615981646589
// MI455X (gfx1250) — matrix operand order repaired and checked by reading, not compiled, not run on hardware
//
#include <hip/hip_runtime.h>
#include <math.h>

// ---- problem sizes (from reference) ----
#define ZH_DIMS 512
#define ZH_HID  1024
#define ZH_VOC  32000
#define EN_HID  512
#define NB      64      // batch
#define NL      32      // seq len
#define EN_LEN  40
#define NG      4096    // 4*ZH_HID

typedef __attribute__((ext_vector_type(16))) _Float16 v16h;
typedef __attribute__((ext_vector_type(8)))  _Float16 v8h;
typedef __attribute__((ext_vector_type(8)))  float    v8f;

// ---------------------------------------------------------------------------
// WMMA fragment helpers (v_wmma_f32_16x16x32_f16; this code's contiguous k order)
// A (16x32 f16, MxK): lane l -> row m = l&15; hi = l>>4;
//   elements j = K[hi*16 + j]
// B (32x16 f16, KxN): lane l -> col n = l&15; elements j = K[hi*16 + j]
// C/D (16x16 f32): lane l -> n = l&15; vgpr r -> m = hi*8 + r
// ---------------------------------------------------------------------------
__device__ __forceinline__ v16h load_a_frag(const _Float16* __restrict__ arow,
                                            int kk, int hi) {
  const _Float16* p = arow + kk + hi * 16;
  v8h lo = *(const v8h*)(p);
  v8h hh = *(const v8h*)(p + 8);
  v16h a;
#pragma unroll
  for (int j = 0; j < 8; ++j) { a[j] = lo[j]; a[j + 8] = hh[j]; }
  return a;
}

__device__ __forceinline__ v16h load_b_frag(const _Float16* __restrict__ B,
                                            int ldb, int n, int kk, int hi) {
  return *(const v16h*)(B + (size_t)n * ldb + kk + hi * 16);
}

__device__ __forceinline__ v8f wmma_f16(v16h a, v16h b, v8f c) {
  return __builtin_amdgcn_wmma_f32_16x16x32_f16(false, a, false, b,
                                                (short)0, c, false, false);
}

// ---------------------------------------------------------------------------
// Software-pipelined (double-buffered) K-loop accumulating 4 M-tiles x 1 N-tile.
// Loads for chunk kk+32 are issued BEFORE WMMAs of chunk kk, so the compiler
// emits partial s_wait_loadcnt instead of wait-to-zero before each WMMA.
// A rows: Abase + (mt*16+lo)*astride ; B rows: Bbase + n*ldb.
// ---------------------------------------------------------------------------
__device__ __forceinline__ void gemm_acc_pipe(v8f* acc,
                                              const _Float16* __restrict__ Abase,
                                              size_t astride,
                                              const _Float16* __restrict__ B,
                                              int ldb, int n, int K,
                                              int hi, int lo) {
  const _Float16* ar[4];
#pragma unroll
  for (int mt = 0; mt < 4; ++mt) ar[mt] = Abase + (size_t)(mt * 16 + lo) * astride;

  v16h bc = load_b_frag(B, ldb, n, 0, hi);
  v16h ac[4];
#pragma unroll
  for (int mt = 0; mt < 4; ++mt) ac[mt] = load_a_frag(ar[mt], 0, hi);

  for (int kk = 32; kk < K; kk += 32) {
    v16h bn = load_b_frag(B, ldb, n, kk, hi);     // prefetch next chunk
    v16h an[4];
#pragma unroll
    for (int mt = 0; mt < 4; ++mt) an[mt] = load_a_frag(ar[mt], kk, hi);
#pragma unroll
    for (int mt = 0; mt < 4; ++mt) acc[mt] = wmma_f16(ac[mt], bc, acc[mt]);
    bc = bn;
#pragma unroll
    for (int mt = 0; mt < 4; ++mt) ac[mt] = an[mt];
  }
#pragma unroll
  for (int mt = 0; mt < 4; ++mt) acc[mt] = wmma_f16(ac[mt], bc, acc[mt]);
}

// ---------------------------------------------------------------------------
// utility kernels
// ---------------------------------------------------------------------------
__global__ void k_f32_to_f16(const float* __restrict__ in,
                             _Float16* __restrict__ out, int n) {
  int i = blockIdx.x * blockDim.x + threadIdx.x;
  int stride = gridDim.x * blockDim.x;
  for (; i < n; i += stride) out[i] = (_Float16)in[i];
}

__global__ void k_zero_f32(float* __restrict__ p, int n) {
  int i = blockIdx.x * blockDim.x + threadIdx.x;
  if (i < n) p[i] = 0.0f;
}

// ---------------------------------------------------------------------------
// hx0 = enc_last[64,512] @ W_p^T[512,1024] + b_p  ->  f16 [64,1024]
// ---------------------------------------------------------------------------
__global__ void k_proj(const _Float16* __restrict__ henc,   // [64,40,512] f16
                       const _Float16* __restrict__ Wp,     // [1024,512] f16
                       const float* __restrict__ bp,
                       _Float16* __restrict__ hx0) {        // [64,1024] f16
  int wave = threadIdx.x >> 5, lane = threadIdx.x & 31;
  int nt = blockIdx.x * 4 + wave;              // 0..63
  int hi = lane >> 4, lo = lane & 15;
  int n = nt * 16 + lo;
  v8f z = {0.f,0.f,0.f,0.f,0.f,0.f,0.f,0.f};
  v8f acc[4] = {z, z, z, z};
  gemm_acc_pipe(acc, henc + (size_t)(EN_LEN - 1) * EN_HID,
                (size_t)EN_LEN * EN_HID, Wp, EN_HID, n, EN_HID, hi, lo);
  float bias = bp[n];
#pragma unroll
  for (int mt = 0; mt < 4; ++mt)
#pragma unroll
    for (int r = 0; r < 8; ++r) {
      int b = mt * 16 + hi * 8 + r;
      hx0[(size_t)b * ZH_HID + n] = (_Float16)(acc[mt][r] + bias);
    }
}

// ---------------------------------------------------------------------------
// Fused LSTM step: one kernel does gates GEMM + cell update.
// Block (128 thr, 4 waves) owns a 16-wide h-slice (ht): wave w computes gate w
// (i,f,g,o) rows n = w*1024 + ht*16 + lo for all 64 batch rows (4 M-tiles),
// K = 512 (x_t @ W_ih^T) then 1024 (hx @ W_hh^T), both pipelined.
// Gate tiles staged in 16KB LDS; then the block performs the elementwise cell
// update for its 64x16 (b,h) patch: fp32 cell state, f16 hx for next GEMMs.
// ---------------------------------------------------------------------------
__global__ void k_step(const _Float16* __restrict__ x,       // [64,32,512] f16
                       const _Float16* __restrict__ hxprev,  // [64,1024] f16
                       const _Float16* __restrict__ Wih,     // [4096,512]
                       const _Float16* __restrict__ Whh,     // [4096,1024]
                       const float* __restrict__ bih,
                       const float* __restrict__ bhh,
                       float* __restrict__ cx,               // [64,1024] f32
                       _Float16* __restrict__ hxnew,         // [64,1024] f16
                       int t) {
  __shared__ float gsm[4 * 1024];              // [gate][b*16 + hl]
  int wave = threadIdx.x >> 5, lane = threadIdx.x & 31;
  int ht = blockIdx.x;                         // 0..63 h-tile
  int hi = lane >> 4, lo = lane & 15;
  int n = wave * ZH_HID + ht * 16 + lo;        // gate row in [0,4096)

  v8f z = {0.f,0.f,0.f,0.f,0.f,0.f,0.f,0.f};
  v8f acc[4] = {z, z, z, z};
  gemm_acc_pipe(acc, x + (size_t)t * ZH_DIMS, (size_t)NL * ZH_DIMS,
                Wih, ZH_DIMS, n, ZH_DIMS, hi, lo);
  gemm_acc_pipe(acc, hxprev, (size_t)ZH_HID,
                Whh, ZH_HID, n, ZH_HID, hi, lo);

  float bias = bih[n] + bhh[n];
#pragma unroll
  for (int mt = 0; mt < 4; ++mt)
#pragma unroll
    for (int r = 0; r < 8; ++r) {
      int b = mt * 16 + hi * 8 + r;
      gsm[wave * 1024 + b * 16 + lo] = acc[mt][r] + bias;
    }
  __syncthreads();

  // cell update: 1024 (b,hl) elements, 128 threads -> 8 each
  for (int e = threadIdx.x; e < 1024; e += 128) {
    int b = e >> 4, hl = e & 15;
    float ig = gsm[e];
    float fg = gsm[1024 + e];
    float gg = gsm[2048 + e];
    float og = gsm[3072 + e];
    float si = 1.0f / (1.0f + __expf(-ig));
    float sf = 1.0f / (1.0f + __expf(-fg));
    float so = 1.0f / (1.0f + __expf(-og));
    size_t ci = (size_t)b * ZH_HID + ht * 16 + hl;
    float c = sf * cx[ci] + si * tanhf(gg);
    cx[ci] = c;
    hxnew[ci] = (_Float16)(so * tanhf(c));
  }
}

// ---------------------------------------------------------------------------
// Batched vocab projection: for each t: logits = hx_t[64,1024] @ W_v^T + b_v
// grid = (ZH_VOC/64, NL); 4 waves/block, each wave: 1 N tile x 4 M tiles.
// Output layout matches reference: logits[b, t, v]
// ---------------------------------------------------------------------------
__global__ void k_logits(const _Float16* __restrict__ hxs,   // [32,64,1024] f16
                         const _Float16* __restrict__ Wv,    // [32000,1024]
                         const float* __restrict__ bv,
                         float* __restrict__ out) {          // [64,32,32000]
  int wave = threadIdx.x >> 5, lane = threadIdx.x & 31;
  int nt = blockIdx.x * 4 + wave;              // 0..1999
  int t  = blockIdx.y;                         // 0..31
  int hi = lane >> 4, lo = lane & 15;
  int n = nt * 16 + lo;
  const _Float16* A = hxs + (size_t)t * NB * ZH_HID;
  v8f z = {0.f,0.f,0.f,0.f,0.f,0.f,0.f,0.f};
  v8f acc[4] = {z, z, z, z};
  gemm_acc_pipe(acc, A, (size_t)ZH_HID, Wv, ZH_HID, n, ZH_HID, hi, lo);
  float bias = bv[n];
#pragma unroll
  for (int mt = 0; mt < 4; ++mt)
#pragma unroll
    for (int r = 0; r < 8; ++r) {
      int b = mt * 16 + hi * 8 + r;
      out[((size_t)b * NL + t) * ZH_VOC + n] = acc[mt][r] + bias;
    }
}

// ---------------------------------------------------------------------------
// argmax over V per (b,t) row; first-occurrence tie-break like jnp.argmax
// ---------------------------------------------------------------------------
__global__ void k_argmax(const float* __restrict__ logits,   // [2048, 32000]
                         float* __restrict__ pred) {         // [2048]
  __shared__ float sv[256];
  __shared__ int   si[256];
  int row = blockIdx.x;
  const float* p = logits + (size_t)row * ZH_VOC;
  float best = -INFINITY; int bidx = 0;
  for (int j = threadIdx.x; j < ZH_VOC; j += 256) {
    float v = p[j];
    if (v > best) { best = v; bidx = j; }
  }
  sv[threadIdx.x] = best; si[threadIdx.x] = bidx;
  __syncthreads();
  for (int s = 128; s > 0; s >>= 1) {
    if (threadIdx.x < (unsigned)s) {
      float v2 = sv[threadIdx.x + s]; int i2 = si[threadIdx.x + s];
      if (v2 > sv[threadIdx.x] ||
          (v2 == sv[threadIdx.x] && i2 < si[threadIdx.x])) {
        sv[threadIdx.x] = v2; si[threadIdx.x] = i2;
      }
    }
    __syncthreads();
  }
  if (threadIdx.x == 0) pred[row] = (float)si[0];
}

// ---------------------------------------------------------------------------
extern "C" void kernel_launch(void* const* d_in, const int* in_sizes, int n_in,
                              void* d_out, int out_size, void* d_ws, size_t ws_size,
                              hipStream_t stream) {
  const float* sent   = (const float*)d_in[0];  // [64,32,512]
  const float* hidden = (const float*)d_in[1];  // [64,40,512]
  const float* W_ih   = (const float*)d_in[2];  // [4096,512]
  const float* W_hh   = (const float*)d_in[3];  // [4096,1024]
  const float* b_ih   = (const float*)d_in[4];
  const float* b_hh   = (const float*)d_in[5];
  const float* W_p    = (const float*)d_in[6];  // [1024,512]
  const float* b_p    = (const float*)d_in[7];
  const float* W_v    = (const float*)d_in[8];  // [32000,1024]
  const float* b_v    = (const float*)d_in[9];

  float* out_logits = (float*)d_out;                             // [64,32,32000]
  float* out_pred   = out_logits + (size_t)NB * NL * ZH_VOC;     // [64,32]

  // bump allocator on workspace (256B aligned)
  char* ws = (char*)d_ws;
  size_t off = 0;
  auto alloc = [&](size_t bytes) -> void* {
    void* p = ws + off;
    off = (off + bytes + 255) & ~(size_t)255;
    return p;
  };
  const size_t n_x    = (size_t)NB * NL * ZH_DIMS;
  const size_t n_henc = (size_t)NB * EN_LEN * EN_HID;
  const size_t n_wp   = (size_t)ZH_HID * EN_HID;
  const size_t n_wih  = (size_t)NG * ZH_DIMS;
  const size_t n_whh  = (size_t)NG * ZH_HID;
  const size_t n_wv   = (size_t)ZH_VOC * ZH_HID;
  _Float16* x_f16    = (_Float16*)alloc(n_x    * 2);
  _Float16* henc_f16 = (_Float16*)alloc(n_henc * 2);
  _Float16* Wp_f16   = (_Float16*)alloc(n_wp   * 2);
  _Float16* Wih_f16  = (_Float16*)alloc(n_wih  * 2);
  _Float16* Whh_f16  = (_Float16*)alloc(n_whh  * 2);
  _Float16* Wv_f16   = (_Float16*)alloc(n_wv   * 2);
  _Float16* hx_all   = (_Float16*)alloc((size_t)(NL + 1) * NB * ZH_HID * 2);
  float*    cx       = (float*)alloc((size_t)NB * ZH_HID * 4);

  // 1) convert weights/activations to f16 once
  k_f32_to_f16<<<2048, 256, 0, stream>>>(sent,   x_f16,    (int)n_x);
  k_f32_to_f16<<<2048, 256, 0, stream>>>(hidden, henc_f16, (int)n_henc);
  k_f32_to_f16<<<1024, 256, 0, stream>>>(W_p,    Wp_f16,   (int)n_wp);
  k_f32_to_f16<<<2048, 256, 0, stream>>>(W_ih,   Wih_f16,  (int)n_wih);
  k_f32_to_f16<<<4096, 256, 0, stream>>>(W_hh,   Whh_f16,  (int)n_whh);
  k_f32_to_f16<<<8192, 256, 0, stream>>>(W_v,    Wv_f16,   (int)n_wv);

  // 2) init state: cx = 0; hx_all[0] = proj(enc_last)
  k_zero_f32<<<(NB * ZH_HID) / 256, 256, 0, stream>>>(cx, NB * ZH_HID);
  k_proj<<<16, 128, 0, stream>>>(henc_f16, Wp_f16, b_p, hx_all);

  // 3) sequential recurrence: one fused kernel per step
  for (int t = 0; t < NL; ++t) {
    _Float16* hx_prev = hx_all + (size_t)t       * NB * ZH_HID;
    _Float16* hx_next = hx_all + (size_t)(t + 1) * NB * ZH_HID;
    k_step<<<64, 128, 0, stream>>>(x_f16, hx_prev, Wih_f16, Whh_f16,
                                   b_ih, b_hh, cx, hx_next, t);
  }

  // 4) one batched vocab GEMM over all 32 steps (W_v streamed once)
  {
    dim3 grid(ZH_VOC / 64, NL);   // (500, 32)
    k_logits<<<grid, 128, 0, stream>>>(hx_all + (size_t)NB * ZH_HID,
                                       Wv_f16, b_v, out_logits);
  }

  // 5) argmax -> predicts[b, t]  (row index b*32+t matches logits layout)
  k_argmax<<<NB * NL, 256, 0, stream>>>(out_logits, out_pred);
}
